// MambaBlock_69243462746256
// MI455X (gfx1250) — matrix operand order repaired and checked by reading, not compiled, not run on hardware
//
#include <hip/hip_runtime.h>
#include <hip/hip_bf16.h>
#include <math.h>

typedef __attribute__((ext_vector_type(16))) _Float16 v16h;
typedef __attribute__((ext_vector_type(8)))  float    v8f;

#define MTOK   16384   // B*L = 4*4096
#define LSEQ   4096
#define DIM_   128
#define DINNER 256
#define NSTATE 16
#define NCHUNK 32
#define CLEN   128     // LSEQ / NCHUNK

// ---------------- fragment loaders (per ISA 16-bit layouts) ----------------

__device__ inline v16h load_a_frag(const _Float16* __restrict__ A, int lda,
                                   int row0, int k0, int lane) {
  int g = lane >> 4, r = lane & 15;
  const _Float16* ap = A + (size_t)(row0 + r) * lda + k0;
  v16h a;
#pragma unroll
  for (int v = 0; v < 8; ++v) {
    int kb = ((v < 4) ? (2 * v) : (2 * v + 8)) + 8 * g;
    a[2 * v]     = ap[kb];
    a[2 * v + 1] = ap[kb + 1];
  }
  return a;
}

// W is [N,K] row-major; B fragment holds B[k][n] = W[n][k]
__device__ inline v16h load_b_frag(const _Float16* __restrict__ W, int ldw,
                                   int col0, int k0, int lane) {
  int g = lane >> 4, n = lane & 15;
  const _Float16* wp = W + (size_t)(col0 + n) * ldw + k0 + 8 * g;
  v16h b;
#pragma unroll
  for (int v = 0; v < 8; ++v) {
    int kb = (v < 4) ? (2 * v) : (2 * v + 8);
    b[2 * v]     = wp[kb];
    b[2 * v + 1] = wp[kb + 1];
  }
  return b;
}

// ---------------- weight conversion ----------------

__global__ void k_f2h(const float* __restrict__ s, _Float16* __restrict__ d, int n) {
  int i = blockIdx.x * blockDim.x + threadIdx.x;
  if (i < n) d[i] = (_Float16)s[i];
}

__global__ void k_f2h_pad(const float* __restrict__ s, _Float16* __restrict__ d,
                          int rows, int rows_pad, int cols) {
  int i = blockIdx.x * blockDim.x + threadIdx.x;
  if (i >= rows_pad * cols) return;
  int r = i / cols;
  d[i] = (r < rows) ? (_Float16)s[i] : (_Float16)0.0f;
}

// ------------- WMMA GEMM, NT column-tiles per wave: O = A[M,K] * W[N,K]^T -------------
// A fragment loaded once per K-step, reused across NT WMMAs (higher wmma:vmem ratio).

template <int NT>
__global__ void k_gemmNT(const _Float16* __restrict__ A, int lda,
                         const _Float16* __restrict__ W, int ldw, int K,
                         float* __restrict__ Of, _Float16* __restrict__ Oh, int ldc,
                         const float* __restrict__ bias) {
  int lane = threadIdx.x & 31;
  int col0 = blockIdx.x * (16 * NT), row0 = blockIdx.y * 16;
  v8f c[NT];
#pragma unroll
  for (int j = 0; j < NT; ++j) { v8f z = {}; c[j] = z; }
  for (int k0 = 0; k0 < K; k0 += 32) {
    v16h a = load_a_frag(A, lda, row0, k0, lane);
#pragma unroll
    for (int j = 0; j < NT; ++j) {
      v16h b = load_b_frag(W, ldw, col0 + 16 * j, k0, lane);
      c[j] = __builtin_amdgcn_wmma_f32_16x16x32_f16(false, a, false, b, (short)0,
                                                    c[j], false, false);
    }
  }
  int g = lane >> 4, n = lane & 15;
#pragma unroll
  for (int j = 0; j < NT; ++j) {
    int col = col0 + 16 * j + n;
    float bv = bias ? bias[col] : 0.0f;
#pragma unroll
    for (int v = 0; v < 8; ++v) {
      int r = row0 + v + 8 * g;
      float val = c[j][v] + bv;
      if (Of) Of[(size_t)r * ldc + col] = val;
      if (Oh) Oh[(size_t)r * ldc + col] = (_Float16)val;
    }
  }
}

// final 1x1 conv: same GEMM but stores into NCHW output [B,128,L] with bias
template <int NT>
__global__ void k_gemmNT_nchw(const _Float16* __restrict__ A, int lda,
                              const _Float16* __restrict__ W, int ldw, int K,
                              float* __restrict__ out, const float* __restrict__ bias) {
  int lane = threadIdx.x & 31;
  int col0 = blockIdx.x * (16 * NT), row0 = blockIdx.y * 16;
  v8f c[NT];
#pragma unroll
  for (int j = 0; j < NT; ++j) { v8f z = {}; c[j] = z; }
  for (int k0 = 0; k0 < K; k0 += 32) {
    v16h a = load_a_frag(A, lda, row0, k0, lane);
#pragma unroll
    for (int j = 0; j < NT; ++j) {
      v16h b = load_b_frag(W, ldw, col0 + 16 * j, k0, lane);
      c[j] = __builtin_amdgcn_wmma_f32_16x16x32_f16(false, a, false, b, (short)0,
                                                    c[j], false, false);
    }
  }
  int g = lane >> 4, n = lane & 15;
#pragma unroll
  for (int j = 0; j < NT; ++j) {
    int col = col0 + 16 * j + n;
    float bv = bias[col];
#pragma unroll
    for (int v = 0; v < 8; ++v) {
      int t = row0 + v + 8 * g;
      int bb = t >> 12, p = t & (LSEQ - 1);
      out[((size_t)bb * DIM_ + col) * LSEQ + p] = c[j][v] + bv;
    }
  }
}

// ---------------- pw_in (NCHW gather) + bias + LayerNorm, fused ----------------
// 1 block = 16 tokens x 128 channels; 2 waves x 4 column tiles; LN via LDS.

__global__ void k_pwin_ln(const float* __restrict__ x, const _Float16* __restrict__ Wh,
                          const float* __restrict__ pb, const float* __restrict__ lng,
                          const float* __restrict__ lnb, _Float16* __restrict__ xn_h) {
  __shared__ float sm[16 * 128];
  __shared__ float smu[16], srs[16];
  int tid = threadIdx.x;
  int lane = tid & 31, wave = tid >> 5;
  int t0 = blockIdx.x * 16;
  int bb = t0 >> 12, p0 = t0 & (LSEQ - 1);
  int colw = wave * 64;
  const float* xb = x + (size_t)bb * DIM_ * LSEQ + p0;
  int g = lane >> 4, r = lane & 15;
  v8f c[4];
#pragma unroll
  for (int j = 0; j < 4; ++j) { v8f z = {}; c[j] = z; }
  for (int k0 = 0; k0 < 128; k0 += 32) {
    v16h a;
#pragma unroll
    for (int v = 0; v < 8; ++v) {
      int kb = ((v < 4) ? (2 * v) : (2 * v + 8)) + 8 * g + k0;
      a[2 * v]     = (_Float16)xb[(size_t)kb * LSEQ + r];
      a[2 * v + 1] = (_Float16)xb[(size_t)(kb + 1) * LSEQ + r];
    }
#pragma unroll
    for (int j = 0; j < 4; ++j) {
      v16h bf = load_b_frag(Wh, 128, colw + 16 * j, k0, lane);
      c[j] = __builtin_amdgcn_wmma_f32_16x16x32_f16(false, a, false, bf, (short)0,
                                                    c[j], false, false);
    }
  }
#pragma unroll
  for (int j = 0; j < 4; ++j) {
    int col = colw + 16 * j + r;
#pragma unroll
    for (int v = 0; v < 8; ++v)
      sm[(v + 8 * g) * 128 + col] = c[j][v] + pb[col];
  }
  __syncthreads();
  if (tid < 16) {
    float s = 0.f, s2 = 0.f;
    for (int cc = 0; cc < 128; ++cc) {
      float vv = sm[tid * 128 + cc];
      s += vv; s2 += vv * vv;
    }
    float mu  = s * (1.0f / 128.0f);
    float var = s2 * (1.0f / 128.0f) - mu * mu;
    smu[tid] = mu;
    srs[tid] = rsqrtf(var + 1e-5f);
  }
  __syncthreads();
  for (int idx = tid; idx < 16 * 128; idx += 64) {
    int rr = idx >> 7, cc = idx & 127;
    float vv = (sm[idx] - smu[rr]) * srs[rr] * lng[cc] + lnb[cc];
    xn_h[(size_t)(t0 + rr) * 128 + cc] = (_Float16)vv;
  }
}

// ---------------- depthwise causal conv1d + SiLU ----------------

__global__ void k_conv_silu(const float* __restrict__ xz, const float* __restrict__ cw,
                            const float* __restrict__ cb,
                            float* __restrict__ ucf, _Float16* __restrict__ uch) {
  int idx = blockIdx.x * blockDim.x + threadIdx.x;  // t*256 + d
  if (idx >= MTOK * DINNER) return;
  int t = idx >> 8, d = idx & 255;
  int l = t & (LSEQ - 1);
  float acc = cb[d];
#pragma unroll
  for (int j = 0; j < 4; ++j) {
    int li = l - 3 + j;
    if (li >= 0) acc += xz[(size_t)(t - 3 + j) * 512 + d] * cw[d * 4 + j];
  }
  float s = acc / (1.0f + __expf(-acc));
  ucf[idx] = s;
  uch[idx] = (_Float16)s;
}

// ---------------- delta = softplus(dt @ W_dt^T + b_dt) ----------------

__global__ void k_delta(const float* __restrict__ xdbl, const float* __restrict__ Wdt,
                        const float* __restrict__ bdt, float* __restrict__ delta) {
  int idx = blockIdx.x * blockDim.x + threadIdx.x;  // t*256 + d
  if (idx >= MTOK * DINNER) return;
  int t = idx >> 8, d = idx & 255;
  const float* dt = xdbl + (size_t)t * 48;
  float acc = bdt[d];
#pragma unroll
  for (int r = 0; r < 8; ++r) acc += dt[r] * Wdt[d * 8 + r];
  delta[idx] = (acc > 20.0f) ? acc : log1pf(__expf(acc));
}

// ---------------- chunked selective scan (3-pass affine decomposition) ----------------
// pass 1: per (b,d,n,chunk): local Q (scan from h=0) and S = sum(delta) -> P = exp(A*S)

__global__ void k_scan1(const float* __restrict__ xdbl, const float* __restrict__ delta,
                        const float* __restrict__ ucf, const float* __restrict__ A_log,
                        float* __restrict__ Qb, float* __restrict__ Sb) {
  int tid = threadIdx.x;
  int ch = blockIdx.x & (NCHUNK - 1);
  int dg = (blockIdx.x >> 5) & 15;
  int bb = blockIdx.x >> 9;
  int d = dg * 16 + (tid >> 4), n = tid & 15;
  float Aneg = -__expf(A_log[d * 16 + n]);
  float S = 0.f, q = 0.f;
  size_t tstart = (size_t)bb * LSEQ + (size_t)ch * CLEN;
  for (int l = 0; l < CLEN; ++l) {
    size_t t = tstart + l;
    float dl = delta[t * 256 + d];
    float ul = ucf[t * 256 + d];
    float Bl = xdbl[t * 48 + 8 + n];
    q = __expf(dl * Aneg) * q + (dl * ul) * Bl;
    S += dl;
  }
  size_t bd = (size_t)bb * 256 + d;
  Qb[(bd * 16 + n) * NCHUNK + ch] = q;
  if (n == 0) Sb[bd * NCHUNK + ch] = S;
}

// pass 2: sequential stitch over chunks: h0[ch] saved, h <- P*h + Q

__global__ void k_scan2(const float* __restrict__ Qb, const float* __restrict__ Sb,
                        const float* __restrict__ A_log, float* __restrict__ H0) {
  int idx = blockIdx.x * blockDim.x + threadIdx.x;  // (b*256+d)*16+n
  if (idx >= 4 * 256 * 16) return;
  int n = idx & 15;
  int d = (idx >> 4) & 255;
  size_t bd = (size_t)(idx >> 4);
  float Aneg = -__expf(A_log[d * 16 + n]);
  float h = 0.0f;
  for (int ch = 0; ch < NCHUNK; ++ch) {
    H0[(size_t)idx * NCHUNK + ch] = h;
    h = __expf(Aneg * Sb[bd * NCHUNK + ch]) * h + Qb[(size_t)idx * NCHUNK + ch];
  }
}

// pass 3: re-run each chunk from its h0, reduce over n, fuse u*D and silu(z) gate

__global__ void k_scan3(const float* __restrict__ xdbl, const float* __restrict__ delta,
                        const float* __restrict__ ucf, const float* __restrict__ xz,
                        const float* __restrict__ A_log, const float* __restrict__ Dp,
                        const float* __restrict__ H0, _Float16* __restrict__ yh) {
  int tid = threadIdx.x;
  int ch = blockIdx.x & (NCHUNK - 1);
  int dg = (blockIdx.x >> 5) & 15;
  int bb = blockIdx.x >> 9;
  int d = dg * 16 + (tid >> 4), n = tid & 15;
  float Aneg = -__expf(A_log[d * 16 + n]);
  float Dv = Dp[d];
  size_t bd = (size_t)bb * 256 + d;
  float h = H0[(bd * 16 + n) * NCHUNK + ch];
  size_t tstart = (size_t)bb * LSEQ + (size_t)ch * CLEN;
  for (int l = 0; l < CLEN; ++l) {
    size_t t = tstart + l;
    if (l + 1 < CLEN && n == 0) {  // prefetch next step (global_prefetch_b8)
      __builtin_prefetch(&delta[(t + 1) * 256 + d], 0, 3);
      __builtin_prefetch(&xdbl[(t + 1) * 48 + 8], 0, 3);
    }
    float Bl = xdbl[t * 48 + 8 + n];
    float Cl = xdbl[t * 48 + 24 + n];
    float dl = delta[t * 256 + d];
    float ul = ucf[t * 256 + d];
    h = __expf(dl * Aneg) * h + (dl * ul) * Bl;
    float contrib = h * Cl;
    contrib += __shfl_xor(contrib, 1, 16);
    contrib += __shfl_xor(contrib, 2, 16);
    contrib += __shfl_xor(contrib, 4, 16);
    contrib += __shfl_xor(contrib, 8, 16);
    if (n == 0) {
      float y = contrib + ul * Dv;
      float z = xz[t * 512 + 256 + d];
      y *= z / (1.0f + __expf(-z));
      yh[t * 256 + d] = (_Float16)y;
    }
  }
}

// ---------------- host launch ----------------

extern "C" void kernel_launch(void* const* d_in, const int* in_sizes, int n_in,
                              void* d_out, int out_size, void* d_ws, size_t ws_size,
                              hipStream_t stream) {
  (void)in_sizes; (void)n_in; (void)out_size; (void)ws_size;
  const float* x      = (const float*)d_in[0];
  const float* pw_in  = (const float*)d_in[1];
  const float* pb_in  = (const float*)d_in[2];
  const float* ln_g   = (const float*)d_in[3];
  const float* ln_b   = (const float*)d_in[4];
  const float* W_in   = (const float*)d_in[5];
  const float* conv_w = (const float*)d_in[6];
  const float* conv_b = (const float*)d_in[7];
  const float* W_x    = (const float*)d_in[8];
  const float* W_dt   = (const float*)d_in[9];
  const float* b_dt   = (const float*)d_in[10];
  const float* A_log  = (const float*)d_in[11];
  const float* Dp     = (const float*)d_in[12];
  const float* W_out  = (const float*)d_in[13];
  const float* pw_out = (const float*)d_in[14];
  const float* pb_out = (const float*)d_in[15];
  float* out = (float*)d_out;

  char* base = (char*)d_ws;
  size_t off = 0;
  auto alloc = [&](size_t bytes) -> char* {
    char* p = base + off;
    off += (bytes + 255) & ~(size_t)255;
    return p;
  };

  _Float16* pw_in_h = (_Float16*)alloc(128 * 128 * 2);
  _Float16* W_in_h  = (_Float16*)alloc(512 * 128 * 2);
  _Float16* W_x_h   = (_Float16*)alloc(48 * 256 * 2);   // padded 40 -> 48 rows
  _Float16* W_out_h = (_Float16*)alloc(128 * 256 * 2);
  _Float16* pw_out_h= (_Float16*)alloc(128 * 128 * 2);
  _Float16* xn_h    = (_Float16*)alloc((size_t)MTOK * 128 * 2);
  float*    xz_f    = (float*)   alloc((size_t)MTOK * 512 * 4);
  _Float16* uc_h    = (_Float16*)alloc((size_t)MTOK * 256 * 2);
  float*    uc_f    = (float*)   alloc((size_t)MTOK * 256 * 4);
  float*    xdbl_f  = (float*)   alloc((size_t)MTOK * 48 * 4);
  float*    delta_f = (float*)   alloc((size_t)MTOK * 256 * 4);
  _Float16* y_h     = (_Float16*)alloc((size_t)MTOK * 256 * 2);
  _Float16* yo_h    = (_Float16*)alloc((size_t)MTOK * 128 * 2);
  float*    Qb      = (float*)   alloc((size_t)4 * 256 * 16 * NCHUNK * 4);
  float*    Sb      = (float*)   alloc((size_t)4 * 256 * NCHUNK * 4);
  float*    H0      = (float*)   alloc((size_t)4 * 256 * 16 * NCHUNK * 4);

  // weight conversions
  k_f2h<<<(128 * 128 + 255) / 256, 256, 0, stream>>>(pw_in, pw_in_h, 128 * 128);
  k_f2h<<<(512 * 128 + 255) / 256, 256, 0, stream>>>(W_in, W_in_h, 512 * 128);
  k_f2h_pad<<<(48 * 256 + 255) / 256, 256, 0, stream>>>(W_x, W_x_h, 40, 48, 256);
  k_f2h<<<(128 * 256 + 255) / 256, 256, 0, stream>>>(W_out, W_out_h, 128 * 256);
  k_f2h<<<(128 * 128 + 255) / 256, 256, 0, stream>>>(pw_out, pw_out_h, 128 * 128);

  // 1) pw_in + bias + LayerNorm -> xn_h  [M,128] f16
  k_pwin_ln<<<MTOK / 16, 64, 0, stream>>>(x, pw_in_h, pb_in, ln_g, ln_b, xn_h);

  // 2) in_proj: [M,128]x[512,128]^T -> xz_f [M,512]
  k_gemmNT<4><<<dim3(512 / 64, MTOK / 16), 32, 0, stream>>>(
      xn_h, 128, W_in_h, 128, 128, xz_f, (_Float16*)nullptr, 512, (const float*)nullptr);

  // 3) depthwise causal conv + SiLU on u (cols 0..255 of xz)
  k_conv_silu<<<(MTOK * DINNER + 255) / 256, 256, 0, stream>>>(xz_f, conv_w, conv_b,
                                                               uc_f, uc_h);

  // 4) x_proj: [M,256]x[48,256]^T -> xdbl [M,48] (cols 40..47 are zero pad)
  k_gemmNT<3><<<dim3(1, MTOK / 16), 32, 0, stream>>>(
      uc_h, 256, W_x_h, 256, 256, xdbl_f, (_Float16*)nullptr, 48, (const float*)nullptr);

  // 5) delta = softplus(dt @ W_dt^T + b_dt)
  k_delta<<<(MTOK * DINNER + 255) / 256, 256, 0, stream>>>(xdbl_f, W_dt, b_dt, delta_f);

  // 6) chunked selective scan + u*D + silu(z) gate -> y_h [M,256] f16
  k_scan1<<<4 * 16 * NCHUNK, 256, 0, stream>>>(xdbl_f, delta_f, uc_f, A_log, Qb, Sb);
  k_scan2<<<(4 * 256 * 16 + 255) / 256, 256, 0, stream>>>(Qb, Sb, A_log, H0);
  k_scan3<<<4 * 16 * NCHUNK, 256, 0, stream>>>(xdbl_f, delta_f, uc_f, xz_f, A_log, Dp,
                                               H0, y_h);

  // 7) out_proj: [M,256]x[128,256]^T -> yo_h [M,128] f16
  k_gemmNT<4><<<dim3(128 / 64, MTOK / 16), 32, 0, stream>>>(
      y_h, 256, W_out_h, 256, 256, (float*)nullptr, yo_h, 128, (const float*)nullptr);

  // 8) pw_out + bias, store NCHW into d_out
  k_gemmNT_nchw<4><<<dim3(128 / 64, MTOK / 16), 32, 0, stream>>>(
      yo_h, 128, pw_out_h, 128, 128, out, pb_out);
}
